// AttentionBlock_816043786289
// MI455X (gfx1250) — compile-verified, matrix operand order checked by reading, not run on hardware
//
#include <hip/hip_runtime.h>
#include <hip/hip_bf16.h>

#define B_ 4
#define C_ 256
#define N_ 4096
#define G_ 32
#define SCALE_ 0.0625f   // C^-0.5

typedef _Float16 half16v __attribute__((ext_vector_type(16)));
typedef _Float16 half8v  __attribute__((ext_vector_type(8)));
typedef float    float8v __attribute__((ext_vector_type(8)));
typedef int      int4v   __attribute__((ext_vector_type(4)));

// ---- gfx1250 async global->LDS path (guarded; sync fallback always compiles)
#if defined(__has_builtin)
#  if __has_builtin(__builtin_amdgcn_global_load_async_to_lds_b128)
#    define USE_ASYNC 1
#  endif
#endif
#ifndef USE_ASYNC
#  define USE_ASYNC 0
#endif

#if USE_ASYNC
typedef __attribute__((address_space(3))) int4v* lds_b128_t;
typedef __attribute__((address_space(1))) int4v* glob_b128_t;   // non-const per builtin signature
#endif

__device__ __forceinline__ void cp16_g2l(const _Float16* gsrc, _Float16* ldst) {
#if USE_ASYNC
  __builtin_amdgcn_global_load_async_to_lds_b128(
      (glob_b128_t)(unsigned long long)gsrc,
      (lds_b128_t)(unsigned int)(unsigned long long)ldst, 0, 0);
#else
  *(int4v*)ldst = *(const int4v*)gsrc;
#endif
}

__device__ __forceinline__ void async_wait_all() {
#if USE_ASYNC
#  if defined(__has_builtin) && __has_builtin(__builtin_amdgcn_s_wait_asynccnt)
  __builtin_amdgcn_s_wait_asynccnt(0);
#  else
  asm volatile("s_wait_asynccnt 0x0" ::: "memory");
#  endif
#endif
}

// Gather one WMMA 16x16x32 f16 operand (A rows / B columns) from a K-contiguous
// row-major source (global or LDS). Lane L<16 holds K {k0+0..7, k0+16..23};
// L>=16 holds K {k0+8..15, k0+24..31} (CDNA5 16-bit matrix VGPR layout).
__device__ __forceinline__ half16v ld_op(const _Float16* rowbase, int stride,
                                         int k0, int lane) {
  int r = lane & 15, h = lane >> 4;
  const _Float16* p = rowbase + (size_t)r * stride + k0 + h * 8;
  half8v lo = *(const half8v*)(p);
  half8v hi = *(const half8v*)(p + 16);
  half16v o;
#pragma unroll
  for (int i = 0; i < 8; ++i) { o[i] = lo[i]; o[i + 8] = hi[i]; }
  return o;
}

__device__ __forceinline__ float8v wmma_f16(half16v a, half16v b, float8v c) {
  return __builtin_amdgcn_wmma_f32_16x16x32_f16(false, a, false, b, (short)0, c,
                                                false, false);
}

// ---------------- weight fp32 -> fp16 (4 matrices, packed q,k,v,p) ----------
__global__ void wconv_kernel(const float* __restrict__ wq, const float* __restrict__ wk,
                             const float* __restrict__ wv, const float* __restrict__ wp,
                             _Float16* __restrict__ w16) {
  int idx = blockIdx.x * 256 + threadIdx.x;      // 0 .. 4*65536-1
  int m = idx >> 16, off = idx & 65535;
  const float* src = (m == 0) ? wq : (m == 1) ? wk : (m == 2) ? wv : wp;
  w16[idx] = (_Float16)src[off];
}

// ---------------- GroupNorm statistics: one block per (b,g) -----------------
__global__ void gn_stats_kernel(const float* __restrict__ x, float* __restrict__ stats) {
  __shared__ float ssum[256], ssq[256];
  int tid = threadIdx.x;
  size_t base = (size_t)blockIdx.x * 32768;      // group data is contiguous
  float s = 0.f, q = 0.f;
  for (int i = tid; i < 32768; i += 256) {
    float v = x[base + i];
    s += v; q += v * v;
  }
  ssum[tid] = s; ssq[tid] = q;
  __syncthreads();
  for (int off = 128; off > 0; off >>= 1) {
    if (tid < off) { ssum[tid] += ssum[tid + off]; ssq[tid] += ssq[tid + off]; }
    __syncthreads();
  }
  if (tid == 0) {
    float mu = ssum[0] * (1.f / 32768.f);
    float var = ssq[0] * (1.f / 32768.f) - mu * mu;
    stats[blockIdx.x * 2 + 0] = mu;
    stats[blockIdx.x * 2 + 1] = rsqrtf(var + 1e-6f);
  }
}

// ---------------- GroupNorm apply, write s^T fp16 [b][n][c] -----------------
__global__ void gn_apply_kernel(const float* __restrict__ x, const float* __restrict__ gamma,
                                const float* __restrict__ beta, const float* __restrict__ stats,
                                _Float16* __restrict__ sT) {
  int n = blockIdx.x * 256 + threadIdx.x;
  int c = blockIdx.y;
  int b = blockIdx.z;
  int g = c >> 3;
  float mu = stats[(b * G_ + g) * 2 + 0];
  float rs = stats[(b * G_ + g) * 2 + 1];
  float v = (x[((size_t)b * C_ + c) * N_ + n] - mu) * rs * gamma[c] + beta[c];
  sT[((size_t)b * N_ + n) * C_ + c] = (_Float16)v;
}

// ------- GEMM: outT[n][o] = sT[n][:] . w[o][:] + bias[o], 16n x 64o per wave
__global__ void __launch_bounds__(128)
gemm_nt_kernel(const _Float16* __restrict__ sT, const _Float16* __restrict__ w16,
               const float* __restrict__ bias, _Float16* __restrict__ outT) {
  int wv = threadIdx.x >> 5, lane = threadIdx.x & 31;
  int wid = blockIdx.x * 4 + wv;                 // 0..1023
  int nt = wid & 255, og = wid >> 8;             // og: group of 4 o-tiles
  int b = blockIdx.y;
  const _Float16* Ab = sT + ((size_t)b * N_ + nt * 16) * C_;
  float8v zero8 = {0.f, 0.f, 0.f, 0.f, 0.f, 0.f, 0.f, 0.f};
  float8v acc[4];
#pragma unroll
  for (int t = 0; t < 4; ++t) acc[t] = zero8;
#pragma unroll
  for (int k0 = 0; k0 < C_; k0 += 32) {
    half16v a = ld_op(Ab, C_, k0, lane);        // shared across 4 o-tiles
#pragma unroll
    for (int t = 0; t < 4; ++t) {
      half16v bo = ld_op(w16 + (size_t)(og * 4 + t) * 16 * C_, C_, k0, lane);
      acc[t] = wmma_f16(a, bo, acc[t]);
    }
  }
  int r = lane & 15, h = lane >> 4;
#pragma unroll
  for (int t = 0; t < 4; ++t) {
    int o = (og * 4 + t) * 16 + r;
    float bvv = bias[o];
#pragma unroll
    for (int e = 0; e < 8; ++e) {
      int n = nt * 16 + e + 8 * h;
      outT[((size_t)b * N_ + n) * C_ + o] = (_Float16)(acc[t][e] + bvv);
    }
  }
}

// ------- GEMM: v[o][n] = wv[o][:] . sT[n][:] + bias[o], 64o x 16n per wave --
__global__ void __launch_bounds__(128)
gemm_v_kernel(const _Float16* __restrict__ w16, const _Float16* __restrict__ sT,
              const float* __restrict__ bias, _Float16* __restrict__ vO) {
  int wv = threadIdx.x >> 5, lane = threadIdx.x & 31;
  int wid = blockIdx.x * 4 + wv;
  int nt = wid & 255, og = wid >> 8;
  int b = blockIdx.y;
  const _Float16* Bb = sT + ((size_t)b * N_ + nt * 16) * C_;   // B cols = sT rows n
  float8v zero8 = {0.f, 0.f, 0.f, 0.f, 0.f, 0.f, 0.f, 0.f};
  float8v acc[4];
#pragma unroll
  for (int t = 0; t < 4; ++t) acc[t] = zero8;
#pragma unroll
  for (int k0 = 0; k0 < C_; k0 += 32) {
    half16v bo = ld_op(Bb, C_, k0, lane);       // shared across 4 o-tiles
#pragma unroll
    for (int t = 0; t < 4; ++t) {
      half16v a = ld_op(w16 + (size_t)(og * 4 + t) * 16 * C_, C_, k0, lane);
      acc[t] = wmma_f16(a, bo, acc[t]);
    }
  }
  int r = lane & 15, h = lane >> 4;
#pragma unroll
  for (int t = 0; t < 4; ++t) {
#pragma unroll
    for (int e = 0; e < 8; ++e) {
      int o = (og * 4 + t) * 16 + e + 8 * h;
      vO[((size_t)b * C_ + o) * N_ + nt * 16 + r] = (_Float16)(acc[t][e] + bias[o]);
    }
  }
}

// ---------------- Flash attention: 4 waves/block, shared async K/V tiles ----
__global__ void __launch_bounds__(128)
attn_kernel(const _Float16* __restrict__ qT, const _Float16* __restrict__ kT,
            const _Float16* __restrict__ vO, _Float16* __restrict__ hT) {
  __shared__ _Float16 sK[2][32][256];   // K tile, double buffered (2 x 16KB)
  __shared__ _Float16 sV[2][256][32];   // V tile, double buffered (2 x 16KB)
  __shared__ float    sS[4][16][32];
  __shared__ _Float16 sP[4][16][32];
  __shared__ float    sM[4][16], sL[4][16], sF[4][16];

  int tid = threadIdx.x;
  int w = tid >> 5, lane = tid & 31;
  int r = lane & 15, hh = lane >> 4;
  int b = blockIdx.y;
  int i0 = (blockIdx.x * 4 + w) * 16;
  const _Float16* qb = qT + ((size_t)b * N_ + i0) * C_;
  const _Float16* kB = kT + (size_t)b * N_ * C_;
  const _Float16* vB = vO + (size_t)b * C_ * N_;

  if (lane < 16) { sM[w][lane] = -3.0e38f; sL[w][lane] = 0.f; }

  // Q operands pinned in registers for the whole kernel (8 x v16h = 64 VGPRs).
  half16v qreg[8];
#pragma unroll
  for (int k = 0; k < 8; ++k) qreg[k] = ld_op(qb, C_, k * 32, lane);

  float8v zero8 = {0.f, 0.f, 0.f, 0.f, 0.f, 0.f, 0.f, 0.f};
  float8v hacc[16];
#pragma unroll
  for (int cc = 0; cc < 16; ++cc) hacc[cc] = zero8;

  // Cooperative copy of one 32-key K tile + V tile: 16 x b128 per thread.
  auto copy_tile = [&](int j0, int buf) {
#pragma unroll
    for (int t = 0; t < 8; ++t) {
      int i = tid + t * 128;                 // 0..1023
      int kr = i >> 5, kc = i & 31;          // K: [32][256] halves
      cp16_g2l(kB + (size_t)(j0 + kr) * C_ + kc * 8, &sK[buf][kr][kc * 8]);
    }
#pragma unroll
    for (int t = 0; t < 8; ++t) {
      int i = tid + t * 128;
      int vr = i >> 2, vc = i & 3;           // V: [256][32] halves
      cp16_g2l(vB + (size_t)vr * N_ + j0 + vc * 8, &sV[buf][vr][vc * 8]);
    }
  };

  copy_tile(0, 0);
  async_wait_all();
  __syncthreads();

  for (int jc = 0; jc < N_ / 32; ++jc) {
    int buf = jc & 1;
    if (jc + 1 < N_ / 32) copy_tile((jc + 1) * 32, buf ^ 1);  // prefetch next

    // ---- scores for 32 keys: two 16x16 tiles, K = C = 256, B from LDS ----
    float8v s0 = zero8, s1 = zero8;
#pragma unroll
    for (int k = 0; k < 8; ++k) {
      half16v b0 = ld_op(&sK[buf][0][0], 256, k * 32, lane);
      half16v b1 = ld_op(&sK[buf][16][0], 256, k * 32, lane);
      s0 = wmma_f16(qreg[k], b0, s0);
      s1 = wmma_f16(qreg[k], b1, s1);
    }
#pragma unroll
    for (int e = 0; e < 8; ++e) {
      int row = e + 8 * hh;
      sS[w][row][r]      = s0[e] * SCALE_;
      sS[w][row][r + 16] = s1[e] * SCALE_;
    }
    __syncthreads();

    // ---- online softmax over the 32 new columns (one row per lane<16) ----
    if (lane < 16) {
      int row = lane;
      float mold = sM[w][row];
      float mx = mold;
#pragma unroll
      for (int j = 0; j < 32; ++j) mx = fmaxf(mx, sS[w][row][j]);
      float fac = __expf(mold - mx);
      float sum = 0.f;
#pragma unroll
      for (int j = 0; j < 32; ++j) {
        float p = __expf(sS[w][row][j] - mx);
        sum += p;
        sP[w][row][j] = (_Float16)p;
      }
      sM[w][row] = mx;
      sL[w][row] = sL[w][row] * fac + sum;
      sF[w][row] = fac;
    }
    __syncthreads();

    // ---- rescale accumulators, then PV: 16 WMMAs across C = 256 ----
    float fv[8];
#pragma unroll
    for (int e = 0; e < 8; ++e) fv[e] = sF[w][hh * 8 + e];

    half8v plo = *(const half8v*)&sP[w][r][hh * 8];
    half8v phi = *(const half8v*)&sP[w][r][16 + hh * 8];
    half16v pa;
#pragma unroll
    for (int i = 0; i < 8; ++i) { pa[i] = plo[i]; pa[i + 8] = phi[i]; }

#pragma unroll
    for (int cc = 0; cc < 16; ++cc) {
#pragma unroll
      for (int e = 0; e < 8; ++e) hacc[cc][e] *= fv[e];
      half16v bv = ld_op(&sV[buf][cc * 16][0], 32, 0, lane);
      hacc[cc] = wmma_f16(pa, bv, hacc[cc]);
    }

    async_wait_all();    // next tile landed in LDS
    __syncthreads();     // all waves done with current buffers
  }

  // ---- epilogue: divide by l, store h^T fp16 [b][n][c] ----
  if (lane < 16) sF[w][lane] = 1.f / sL[w][lane];
  __syncthreads();
  float inv[8];
#pragma unroll
  for (int e = 0; e < 8; ++e) inv[e] = sF[w][hh * 8 + e];
#pragma unroll
  for (int cc = 0; cc < 16; ++cc) {
#pragma unroll
    for (int e = 0; e < 8; ++e) {
      int n = i0 + e + 8 * hh;
      hT[((size_t)b * N_ + n) * C_ + cc * 16 + r] = (_Float16)(hacc[cc][e] * inv[e]);
    }
  }
}

// ---------------- Projection + bias + residual, f32 out [b][o][n] -----------
__global__ void __launch_bounds__(128)
proj_kernel(const _Float16* __restrict__ hT, const _Float16* __restrict__ w16,
            const float* __restrict__ bias, const float* __restrict__ x,
            float* __restrict__ out) {
  int wv = threadIdx.x >> 5, lane = threadIdx.x & 31;
  int wid = blockIdx.x * 4 + wv;
  int nt = wid & 255, og = wid >> 8;
  int b = blockIdx.y;
  const _Float16* Ab = hT + ((size_t)b * N_ + nt * 16) * C_;
  float8v zero8 = {0.f, 0.f, 0.f, 0.f, 0.f, 0.f, 0.f, 0.f};
  float8v acc[4];
#pragma unroll
  for (int t = 0; t < 4; ++t) acc[t] = zero8;
#pragma unroll
  for (int k0 = 0; k0 < C_; k0 += 32) {
    half16v a = ld_op(Ab, C_, k0, lane);
#pragma unroll
    for (int t = 0; t < 4; ++t) {
      half16v bo = ld_op(w16 + (size_t)(og * 4 + t) * 16 * C_, C_, k0, lane);
      acc[t] = wmma_f16(a, bo, acc[t]);
    }
  }
  int r = lane & 15, h = lane >> 4;
#pragma unroll
  for (int t = 0; t < 4; ++t) {
    int o = (og * 4 + t) * 16 + r;
    float bvv = bias[o];
#pragma unroll
    for (int e = 0; e < 8; ++e) {
      int n = nt * 16 + e + 8 * h;
      size_t idx = ((size_t)b * C_ + o) * N_ + n;
      out[idx] = x[idx] + acc[t][e] + bvv;
    }
  }
}

extern "C" void kernel_launch(void* const* d_in, const int* in_sizes, int n_in,
                              void* d_out, int out_size, void* d_ws, size_t ws_size,
                              hipStream_t stream) {
  (void)in_sizes; (void)n_in; (void)out_size; (void)ws_size;
  const float* x     = (const float*)d_in[0];
  const float* gamma = (const float*)d_in[1];
  const float* beta  = (const float*)d_in[2];
  const float* wq    = (const float*)d_in[3];
  const float* bq    = (const float*)d_in[4];
  const float* wk    = (const float*)d_in[5];
  const float* bk    = (const float*)d_in[6];
  const float* wv    = (const float*)d_in[7];
  const float* bv    = (const float*)d_in[8];
  const float* wp    = (const float*)d_in[9];
  const float* bp    = (const float*)d_in[10];
  float* out = (float*)d_out;

  char* ws = (char*)d_ws;
  _Float16* w16  = (_Float16*)ws;                        // 4 * 256*256 fp16
  float*    stats = (float*)(ws + 524288);               // 128 * {mu, rstd}
  _Float16* sT = (_Float16*)(ws + 525312);               // [B][N][C] fp16
  const size_t PL = (size_t)B_ * N_ * C_;
  _Float16* qT = sT + PL;
  _Float16* kT = qT + PL;
  _Float16* vO = kT + PL;
  _Float16* hT = vO + PL;

  wconv_kernel<<<1024, 256, 0, stream>>>(wq, wk, wv, wp, w16);
  gn_stats_kernel<<<B_ * G_, 256, 0, stream>>>(x, stats);
  gn_apply_kernel<<<dim3(N_ / 256, C_, B_), 256, 0, stream>>>(x, gamma, beta, stats, sT);
  gemm_nt_kernel<<<dim3(256, B_), 128, 0, stream>>>(sT, w16, bq, qT);
  gemm_nt_kernel<<<dim3(256, B_), 128, 0, stream>>>(sT, w16 + 65536, bk, kT);
  gemm_v_kernel<<<dim3(256, B_), 128, 0, stream>>>(w16 + 2 * 65536, sT, bv, vO);
  attn_kernel<<<dim3(N_ / 64, B_), 128, 0, stream>>>(qT, kT, vO, hT);
  proj_kernel<<<dim3(256, B_), 128, 0, stream>>>(hT, w16 + 3 * 65536, bp, x, out);
}
